// NATTEN_20761871909088
// MI455X (gfx1250) — matrix operand order repaired and checked by reading, not compiled, not run on hardware
//
#include <hip/hip_runtime.h>

// ---------------------------------------------------------------------------
// Types for CDNA5 WMMA (wave32): v_wmma_f32_16x16x32_f16
// ---------------------------------------------------------------------------
typedef __attribute__((ext_vector_type(16))) _Float16 v16h;
typedef __attribute__((ext_vector_type(8)))  _Float16 v8h;
typedef __attribute__((ext_vector_type(8)))  float    v8f;

union V16U { v16h v; v8h h[2]; };

#define T_DIM 256
#define P_DIM 88
#define C_DIM 128
#define N_HEADS 4
#define HD 32
#define WIN 25
#define TP (T_DIM * P_DIM)   /* 22528 */
#define RPB_W 49
#define VTP TP               /* v_t plane pitch */

__device__ __forceinline__ v8f wmma16(v16h a, v16h b, v8f c) {
  // D = A(16x32 f16) * B(32x16 f16) + C(16x16 f32)
  return __builtin_amdgcn_wmma_f32_16x16x32_f16(false, a, false, b, (short)0, c,
                                                false, false);
}

__device__ __forceinline__ int iclamp(int v, int lo, int hi) {
  v = v < lo ? lo : v;
  return v > hi ? hi : v;
}

// ---------------------------------------------------------------------------
// Kernel 1: fused concat([x,cond,mask]) @ linear_w^T + b, ReLU  -> y (f16)
// M=22528, K=131 (padded to 160), N=128.  64x64 tile, 4 waves/WG.
// ---------------------------------------------------------------------------
__global__ __launch_bounds__(128)
void fusion_kernel(const float* __restrict__ x, const float* __restrict__ cond,
                   const float* __restrict__ mask, const float* __restrict__ W,
                   const float* __restrict__ bias, _Float16* __restrict__ yh) {
  __shared__ _Float16 As[64 * 48];
  __shared__ _Float16 Ws[64 * 48];
  const int tid  = threadIdx.x;
  const int lane = tid & 31, wave = tid >> 5;
  const int mbase = blockIdx.x * 64;
  const int nbase = blockIdx.y * 64;
  const int srow = tid >> 1, spart = (tid & 1) * 16;
  const int ml = lane & 15, hi = lane >> 4, b8 = hi * 8;

  v8f acc[4] = {{0,0,0,0,0,0,0,0},{0,0,0,0,0,0,0,0},
                {0,0,0,0,0,0,0,0},{0,0,0,0,0,0,0,0}};

  for (int kb = 0; kb < 160; kb += 32) {
    {
      const size_t m = (size_t)(mbase + srow);
      for (int i = 0; i < 16; ++i) {
        const int k = kb + spart + i;
        float v;
        if (k < 128)      v = x[m * 128 + k];
        else if (k == 128) v = cond[m * 2 + 0];
        else if (k == 129) v = cond[m * 2 + 1];
        else if (k == 130) v = mask[m];
        else               v = 0.f;
        As[srow * 48 + spart + i] = (_Float16)v;
      }
      const size_t n = (size_t)(nbase + srow);
      for (int i = 0; i < 16; ++i) {
        const int k = kb + spart + i;
        Ws[srow * 48 + spart + i] = (_Float16)((k < 131) ? W[n * 131 + k] : 0.f);
      }
    }
    __syncthreads();
    V16U A;
    A.h[0] = *(const v8h*)&As[(wave * 16 + ml) * 48 + hi * 16];
    A.h[1] = *(const v8h*)&As[(wave * 16 + ml) * 48 + hi * 16 + 8];
    #pragma unroll
    for (int nb = 0; nb < 4; ++nb) {
      V16U Bf;
      Bf.h[0] = *(const v8h*)&Ws[(nb * 16 + ml) * 48 + hi * 16];
      Bf.h[1] = *(const v8h*)&Ws[(nb * 16 + ml) * 48 + hi * 16 + 8];
      acc[nb] = wmma16(A.v, Bf.v, acc[nb]);
    }
    __syncthreads();
  }
  #pragma unroll
  for (int nb = 0; nb < 4; ++nb)
    #pragma unroll
    for (int r = 0; r < 8; ++r) {
      const int m = mbase + wave * 16 + r + 8 * hi;
      const int n = nbase + nb * 16 + ml;
      float v = acc[nb][r] + bias[n];
      v = v > 0.f ? v : 0.f;
      yh[(size_t)m * C_DIM + n] = (_Float16)v;
    }
}

// ---------------------------------------------------------------------------
// Kernel 2: qkv = y @ qkv_w^T + b.  M=22528, K=128, N=384.
// Epilogue scatters to q (scaled by 32^-0.5), k : [head][t*88+p][32] f16
// and v transposed: v_t [head][hd][t*88+p] f16.
// ---------------------------------------------------------------------------
__global__ __launch_bounds__(128)
void qkv_kernel(const _Float16* __restrict__ yh, const float* __restrict__ W,
                const float* __restrict__ bias, _Float16* __restrict__ qb,
                _Float16* __restrict__ kbuf, _Float16* __restrict__ vt) {
  __shared__ _Float16 As[64 * 48];
  __shared__ _Float16 Ws[64 * 48];
  const int tid  = threadIdx.x;
  const int lane = tid & 31, wave = tid >> 5;
  const int mbase = blockIdx.x * 64;
  const int nbase = blockIdx.y * 64;
  const int srow = tid >> 1, spart = (tid & 1) * 16;
  const int ml = lane & 15, hi = lane >> 4, b8 = hi * 8;

  v8f acc[4] = {{0,0,0,0,0,0,0,0},{0,0,0,0,0,0,0,0},
                {0,0,0,0,0,0,0,0},{0,0,0,0,0,0,0,0}};

  for (int kb = 0; kb < 128; kb += 32) {
    {
      const size_t m = (size_t)(mbase + srow);
      *(v8h*)&As[srow * 48 + spart]     = *(const v8h*)&yh[m * 128 + kb + spart];
      *(v8h*)&As[srow * 48 + spart + 8] = *(const v8h*)&yh[m * 128 + kb + spart + 8];
      const size_t n = (size_t)(nbase + srow);
      for (int i = 0; i < 16; ++i)
        Ws[srow * 48 + spart + i] = (_Float16)W[n * 128 + kb + spart + i];
    }
    __syncthreads();
    V16U A;
    A.h[0] = *(const v8h*)&As[(wave * 16 + ml) * 48 + hi * 16];
    A.h[1] = *(const v8h*)&As[(wave * 16 + ml) * 48 + hi * 16 + 8];
    #pragma unroll
    for (int nb = 0; nb < 4; ++nb) {
      V16U Bf;
      Bf.h[0] = *(const v8h*)&Ws[(nb * 16 + ml) * 48 + hi * 16];
      Bf.h[1] = *(const v8h*)&Ws[(nb * 16 + ml) * 48 + hi * 16 + 8];
      acc[nb] = wmma16(A.v, Bf.v, acc[nb]);
    }
    __syncthreads();
  }
  #pragma unroll
  for (int nb = 0; nb < 4; ++nb)
    #pragma unroll
    for (int r = 0; r < 8; ++r) {
      const int m = mbase + wave * 16 + r + 8 * hi;
      const int n = nbase + nb * 16 + ml;
      const float v = acc[nb][r] + bias[n];
      const int part = n >> 7, c = n & 127, head = c >> 5, hd = c & 31;
      if (part == 0)
        qb[((size_t)head * TP + m) * HD + hd] = (_Float16)(v * 0.17677669529663687f);
      else if (part == 1)
        kbuf[((size_t)head * TP + m) * HD + hd] = (_Float16)v;
      else
        vt[(size_t)(head * HD + hd) * VTP + m] = (_Float16)v;
    }
}

// ---------------------------------------------------------------------------
// Kernel 3: 25x25 edge-clamped neighborhood attention.
// One wave per (j-tile of 16 queries, t, head). Scores: Q(16x32) x K^T per
// 16-key chunk -> +rpb / window mask -> LDS; wave softmax; PV: P(16x32keys)
// x V(32x16hd) from transposed v_t. All fragments are aligned b128 loads.
// ---------------------------------------------------------------------------
__global__ __launch_bounds__(32)
void attn_kernel(const _Float16* __restrict__ q, const _Float16* __restrict__ k,
                 const _Float16* __restrict__ vt, const float* __restrict__ rpb,
                 _Float16* __restrict__ out) {
  __shared__ _Float16 P[16 * 1600];   // [query][kr*64 + kc], 50 KB
  const int lane = threadIdx.x;
  const int ml = lane & 15, hi = lane >> 4;
  const int j0 = blockIdx.x * 16;
  const int t = blockIdx.y;
  const int head = blockIdx.z;

  const int r0  = iclamp(t - 12, 0, T_DIM - WIN);
  const int c0  = iclamp(j0 - 12, 0, P_DIM - WIN);
  const int c0a = c0 & ~7;                       // align col base for b128 V loads
  const int jmax = (j0 + 15 < P_DIM) ? j0 + 15 : P_DIM - 1;
  const int c1  = iclamp(jmax - 12, 0, P_DIM - WIN);
  const int WuA = c1 + WIN - c0a;                // union width (<= 47)
  const int NCH = (WuA + 15) >> 4;               // 2 or 3 chunks of 16 keys
  const int WRIT = NCH << 4;

  // --- Q fragment (A: 16 queries x 32 hd) ---
  const int jq = (j0 + ml < P_DIM) ? j0 + ml : P_DIM - 1;
  const _Float16* qp = q + ((size_t)head * TP + (size_t)t * P_DIM + jq) * HD;
  V16U A;
  A.h[0] = *(const v8h*)(qp + hi * 16);
  A.h[1] = *(const v8h*)(qp + hi * 16 + 8);

  // --- Pass 1: scores + bias + mask -> LDS ---
  for (int kr = 0; kr < WIN; ++kr) {
    const int tk = r0 + kr;
    const int rel_h = tk - t + 24;
    for (int nc = 0; nc < NCH; ++nc) {
      const int kc = nc * 16 + ml;
      const int pkr = c0a + kc;
      const int pk = pkr < P_DIM ? pkr : P_DIM - 1;
      const _Float16* kp = k + ((size_t)head * TP + (size_t)tk * P_DIM + pk) * HD;
      V16U Bf;
      Bf.h[0] = *(const v8h*)(kp + hi * 16);
      Bf.h[1] = *(const v8h*)(kp + hi * 16 + 8);
      v8f S = {0, 0, 0, 0, 0, 0, 0, 0};
      S = wmma16(A.v, Bf.v, S);
      #pragma unroll
      for (int r = 0; r < 8; ++r) {
        const int qr = r + 8 * hi;
        const int j = j0 + qr;
        const int jc = j < P_DIM ? j : P_DIM - 1;
        const int cs = iclamp(jc - 12, 0, P_DIM - WIN);
        float sv;
        if (pkr >= cs && pkr <= cs + 24)
          sv = S[r] + rpb[((size_t)head * RPB_W + rel_h) * RPB_W + (pkr - jc + 24)];
        else
          sv = -1e30f;   // -> -inf in f16, exp() == 0
        P[qr * 1600 + kr * 64 + nc * 16 + ml] = (_Float16)sv;
      }
    }
  }
  __syncthreads();

  // --- Pass 2: softmax (lane -> query ml; lane-halves split the key rows) ---
  {
    float mx = -3.0e38f;
    for (int kr = hi; kr < WIN; kr += 2)
      for (int kc = 0; kc < WRIT; ++kc) {
        const float v = (float)P[ml * 1600 + kr * 64 + kc];
        mx = v > mx ? v : mx;
      }
    const float mo = __shfl_xor(mx, 16, 32);
    mx = mo > mx ? mo : mx;
    float ls = 0.f;
    for (int kr = hi; kr < WIN; kr += 2)
      for (int kc = 0; kc < WRIT; ++kc)
        ls += __expf((float)P[ml * 1600 + kr * 64 + kc] - mx);
    ls += __shfl_xor(ls, 16, 32);
    const float inv = 1.f / ls;
    for (int kr = hi; kr < WIN; kr += 2)
      for (int kc = 0; kc < 64; ++kc) {
        float p = 0.f;
        if (kc < WRIT) p = __expf((float)P[ml * 1600 + kr * 64 + kc] - mx) * inv;
        P[ml * 1600 + kr * 64 + kc] = (_Float16)p;   // pad cols -> 0
      }
  }
  __syncthreads();

  // --- Pass 3: O = P @ V (two 16x16 accumulators -> hd 0..15 / 16..31) ---
  v8f O0 = {0, 0, 0, 0, 0, 0, 0, 0};
  v8f O1 = {0, 0, 0, 0, 0, 0, 0, 0};
  const int nch32 = (WRIT > 32) ? 2 : 1;
  for (int kr = 0; kr < WIN; ++kr) {
    const size_t rowoff = (size_t)(r0 + kr) * P_DIM + c0a;
    for (int cc = 0; cc < nch32; ++cc) {
      const int cb = cc * 32;
      V16U PA;
      const _Float16* pp = &P[ml * 1600 + kr * 64 + cb];
      PA.h[0] = *(const v8h*)(pp + hi * 16);
      PA.h[1] = *(const v8h*)(pp + hi * 16 + 8);
      const _Float16* v0 = vt + (size_t)(head * HD + ml) * VTP + rowoff + cb + hi * 16;
      V16U B0; B0.h[0] = *(const v8h*)v0; B0.h[1] = *(const v8h*)(v0 + 8);
      O0 = wmma16(PA.v, B0.v, O0);
      const _Float16* v1 = vt + (size_t)(head * HD + 16 + ml) * VTP + rowoff + cb + hi * 16;
      V16U B1; B1.h[0] = *(const v8h*)v1; B1.h[1] = *(const v8h*)(v1 + 8);
      O1 = wmma16(PA.v, B1.v, O1);
    }
  }
  #pragma unroll
  for (int r = 0; r < 8; ++r) {
    const int qr = r + 8 * hi;
    const int j = j0 + qr;
    if (j < P_DIM) {
      const size_t row = ((size_t)t * P_DIM + j) * C_DIM + head * HD;
      out[row + ml]      = (_Float16)O0[r];
      out[row + 16 + ml] = (_Float16)O1[r];
    }
  }
}

// ---------------------------------------------------------------------------
// Kernel 4: proj = attn @ proj_w^T + b -> y (f16, next layer) and f32 (d_out)
// ---------------------------------------------------------------------------
__global__ __launch_bounds__(128)
void proj_kernel(const _Float16* __restrict__ a, const float* __restrict__ W,
                 const float* __restrict__ bias, _Float16* __restrict__ yh,
                 float* __restrict__ yf) {
  __shared__ _Float16 As[64 * 48];
  __shared__ _Float16 Ws[64 * 48];
  const int tid  = threadIdx.x;
  const int lane = tid & 31, wave = tid >> 5;
  const int mbase = blockIdx.x * 64;
  const int nbase = blockIdx.y * 64;
  const int srow = tid >> 1, spart = (tid & 1) * 16;
  const int ml = lane & 15, hi = lane >> 4, b8 = hi * 8;

  v8f acc[4] = {{0,0,0,0,0,0,0,0},{0,0,0,0,0,0,0,0},
                {0,0,0,0,0,0,0,0},{0,0,0,0,0,0,0,0}};

  for (int kb = 0; kb < 128; kb += 32) {
    {
      const size_t m = (size_t)(mbase + srow);
      *(v8h*)&As[srow * 48 + spart]     = *(const v8h*)&a[m * 128 + kb + spart];
      *(v8h*)&As[srow * 48 + spart + 8] = *(const v8h*)&a[m * 128 + kb + spart + 8];
      const size_t n = (size_t)(nbase + srow);
      for (int i = 0; i < 16; ++i)
        Ws[srow * 48 + spart + i] = (_Float16)W[n * 128 + kb + spart + i];
    }
    __syncthreads();
    V16U A;
    A.h[0] = *(const v8h*)&As[(wave * 16 + ml) * 48 + hi * 16];
    A.h[1] = *(const v8h*)&As[(wave * 16 + ml) * 48 + hi * 16 + 8];
    #pragma unroll
    for (int nb = 0; nb < 4; ++nb) {
      V16U Bf;
      Bf.h[0] = *(const v8h*)&Ws[(nb * 16 + ml) * 48 + hi * 16];
      Bf.h[1] = *(const v8h*)&Ws[(nb * 16 + ml) * 48 + hi * 16 + 8];
      acc[nb] = wmma16(A.v, Bf.v, acc[nb]);
    }
    __syncthreads();
  }
  #pragma unroll
  for (int nb = 0; nb < 4; ++nb)
    #pragma unroll
    for (int r = 0; r < 8; ++r) {
      const int m = mbase + wave * 16 + r + 8 * hi;
      const int n = nbase + nb * 16 + ml;
      const float v = acc[nb][r] + bias[n];
      yh[(size_t)m * C_DIM + n] = (_Float16)v;
      yf[(size_t)m * C_DIM + n] = v;
    }
}

// ---------------------------------------------------------------------------
// Launch
// ---------------------------------------------------------------------------
extern "C" void kernel_launch(void* const* d_in, const int* in_sizes, int n_in,
                              void* d_out, int out_size, void* d_ws, size_t ws_size,
                              hipStream_t stream) {
  (void)in_sizes; (void)n_in; (void)out_size; (void)ws_size;
  const float* x      = (const float*)d_in[0];
  const float* cond   = (const float*)d_in[1];
  const float* mask   = (const float*)d_in[2];
  const float* lin_w  = (const float*)d_in[3];
  const float* lin_b  = (const float*)d_in[4];
  const float* qkv_w  = (const float*)d_in[5];
  const float* qkv_b  = (const float*)d_in[6];
  const float* rpb    = (const float*)d_in[7];
  const float* proj_w = (const float*)d_in[8];
  const float* proj_b = (const float*)d_in[9];
  float* out = (float*)d_out;

  _Float16* ws = (_Float16*)d_ws;
  const size_t SEG = (size_t)TP * C_DIM;       // 2,883,584 halves
  _Float16* yh   = ws;                         // [TP][128]
  _Float16* qb   = ws + SEG;                   // [4][TP][32]
  _Float16* kb   = ws + 2 * SEG;               // [4][TP][32]
  _Float16* vt   = ws + 3 * SEG;               // [4*32][TP] + 256 slack
  _Float16* attn = ws + 4 * SEG + 256;         // [TP][128]
  // total: 5*SEG + 256 halves ~= 27.5 MB of workspace

  dim3 blk(128);
  fusion_kernel<<<dim3(TP / 64, C_DIM / 64), blk, 0, stream>>>(
      x, cond, mask, lin_w, lin_b, yh);
  for (int layer = 0; layer < 2; ++layer) {
    qkv_kernel<<<dim3(TP / 64, 384 / 64), blk, 0, stream>>>(
        yh, qkv_w, qkv_b, qb, kb, vt);
    attn_kernel<<<dim3(6, T_DIM, N_HEADS), dim3(32), 0, stream>>>(
        qb, kb, vt, rpb, attn);
    proj_kernel<<<dim3(TP / 64, C_DIM / 64), blk, 0, stream>>>(
        attn, proj_w, proj_b, yh, out);
  }
}
